// ContextContrastiveLoss_86268713107950
// MI455X (gfx1250) — matrix operand order repaired and checked by reading, not compiled, not run on hardware
//
#include <hip/hip_runtime.h>
#include <hip/hip_bf16.h>
#include <math.h>

typedef __attribute__((ext_vector_type(16))) _Float16 v16h;
typedef __attribute__((ext_vector_type(8)))  _Float16 v8h;
typedef __attribute__((ext_vector_type(8)))  float    v8f;

#define INV_T 14.285714285714286f   // 1 / 0.07

// ---------------------------------------------------------------------------
// Kernel 1: L2-normalize rows of x [N x 64] (f32) -> z [N x 64] (f16).
// One wave32 per row; lane loads float2 (coalesced), wave reduce via shfl_xor.
// Also zeroes the accumulator buffer acc[0..2N) (expsum then pair counts).
// ---------------------------------------------------------------------------
__global__ void __launch_bounds__(256)
normalize_f16_kernel(const float* __restrict__ x,
                     _Float16* __restrict__ z,
                     float* __restrict__ acc,
                     int N) {
  const int gtid = blockIdx.x * blockDim.x + threadIdx.x;
  if (gtid < 2 * N) acc[gtid] = 0.0f;      // zero expsum + count accumulators

  const int lane = threadIdx.x & 31;
  const int row  = gtid >> 5;
  if (row >= N) return;

  const float2 v = ((const float2*)(x + (size_t)row * 64))[lane];
  float ss = v.x * v.x + v.y * v.y;
  #pragma unroll
  for (int o = 1; o < 32; o <<= 1) ss += __shfl_xor(ss, o, 32);

  const float inv = 1.0f / fmaxf(sqrtf(ss), 1e-12f);
  _Float16* zr = z + (size_t)row * 64;
  zr[2 * lane]     = (_Float16)(v.x * inv);
  zr[2 * lane + 1] = (_Float16)(v.y * inv);
}

// ---------------------------------------------------------------------------
// Kernel 2: For a 16-row strip I, stream J tiles; S-tile = A(16x64) * B(64x16)
// via two v_wmma_f32_16x16x32_f16. Epilogue: mask (same token, j != i),
// accumulate exp(-S/T) and pair count per row; lane-reduce; atomicAdd to acc.
//
// A layout (16x32, same k order as B): lane<16 -> row=lane, K 0..15;
//                                        lane>=16 -> row=lane-16, K 16..31.
// B layout (32x16): lanes 0-15 col=lane K 0..15; lanes 16-31 col=lane-16 K 16..31.
// C/D layout: VGPR r = row (r | r+8 for upper lanes), col = lane&15.
// ---------------------------------------------------------------------------
__global__ void __launch_bounds__(128)
sim_expsum_kernel(const _Float16* __restrict__ z,
                  const int* __restrict__ tok,
                  float* __restrict__ acc,   // [0,N)=expsum, [N,2N)=count
                  int N, int tilesPerChunk) {
  const int lane   = threadIdx.x & 31;
  const int wid    = threadIdx.x >> 5;
  const int I      = blockIdx.x * 16;             // row-strip base
  const int chunk  = blockIdx.y * 4 + wid;        // 16 J-chunks total
  const int lr     = lane & 15;
  const int hs     = lane >> 4;                   // half-wave select
  const int rowbase = I + hs * 8;                 // C/D rows held by this lane

  // ---- load A tile once (K = 0..63 as two 16x32 operands) ----
  const _Float16* za = z + (size_t)(I + lr) * 64 + hs * 16;
  const v8h a0lo = *(const v8h*)(za);        // K  0.. 7 (+hs*16)
  const v8h a0hi = *(const v8h*)(za + 8);    // K  8..15 (+hs*16)
  const v8h a1lo = *(const v8h*)(za + 32);   // K 32..39 (+hs*16)
  const v8h a1hi = *(const v8h*)(za + 40);   // K 40..47 (+hs*16)
  v16h a0, a1;
  #pragma unroll
  for (int i = 0; i < 8; ++i) {
    a0[i] = a0lo[i]; a0[i + 8] = a0hi[i];
    a1[i] = a1lo[i]; a1[i + 8] = a1hi[i];
  }

  int trow[8];
  #pragma unroll
  for (int r = 0; r < 8; ++r) trow[r] = tok[rowbase + r];

  float accS[8], accC[8];
  #pragma unroll
  for (int r = 0; r < 8; ++r) { accS[r] = 0.0f; accC[r] = 0.0f; }

  const int t0 = chunk * tilesPerChunk;
  for (int t = 0; t < tilesPerChunk; ++t) {
    const int J = (t0 + t) * 16;
    const _Float16* zb = z + (size_t)(J + lr) * 64 + hs * 16;
    const v16h b0 = *(const v16h*)(zb);        // K  0..31 operand
    const v16h b1 = *(const v16h*)(zb + 32);   // K 32..63 operand
    __builtin_prefetch((const void*)(zb + 16 * 64), 0, 1);  // next J tile

    v8f c = {};
    c = __builtin_amdgcn_wmma_f32_16x16x32_f16(false, a0, false, b0,
                                               (short)0, c, false, false);
    c = __builtin_amdgcn_wmma_f32_16x16x32_f16(false, a1, false, b1,
                                               (short)0, c, false, false);

    const int col  = J + lr;
    const int tcol = tok[col];
    #pragma unroll
    for (int r = 0; r < 8; ++r) {
      const int  row = rowbase + r;
      const bool m   = (trow[r] == tcol) && (row != col);
      accS[r] += m ? __expf(c[r] * -INV_T) : 0.0f;
      accC[r] += m ? 1.0f : 0.0f;
    }
  }

  // reduce across the 16 lanes that share each row (stay within half-wave)
  #pragma unroll
  for (int r = 0; r < 8; ++r) {
    #pragma unroll
    for (int o = 1; o < 16; o <<= 1) {
      accS[r] += __shfl_xor(accS[r], o, 32);
      accC[r] += __shfl_xor(accC[r], o, 32);
    }
  }
  #pragma unroll
  for (int r = 0; r < 8; ++r) {
    if (lr == r) {
      atomicAdd(&acc[rowbase + r],     accS[r]);
      atomicAdd(&acc[N + rowbase + r], accC[r]);
    }
  }
}

// ---------------------------------------------------------------------------
// Kernel 3: loss = sum_i [cnt_i>0] (-log expsum_i) / sum_i cnt_i  (single block)
// ---------------------------------------------------------------------------
__global__ void __launch_bounds__(256)
finalize_kernel(const float* __restrict__ acc, float* __restrict__ out, int N) {
  __shared__ float sl[256];
  __shared__ float sp[256];
  const int tid = threadIdx.x;
  float l = 0.0f, p = 0.0f;
  for (int i = tid; i < N; i += 256) {
    const float c = acc[N + i];
    if (c > 0.0f) { l -= logf(acc[i]); p += c; }
  }
  sl[tid] = l; sp[tid] = p;
  __syncthreads();
  for (int s = 128; s > 0; s >>= 1) {
    if (tid < s) { sl[tid] += sl[tid + s]; sp[tid] += sp[tid + s]; }
    __syncthreads();
  }
  if (tid == 0) out[0] = (sp[0] > 0.0f) ? (sl[0] / sp[0]) : 0.0f;
}

// ---------------------------------------------------------------------------
extern "C" void kernel_launch(void* const* d_in, const int* in_sizes, int n_in,
                              void* d_out, int out_size, void* d_ws, size_t ws_size,
                              hipStream_t stream) {
  const float* x  = (const float*)d_in[0];   // semantic_state [B,T,64] f32
  const int* tok  = (const int*)d_in[1];     // token_ids [B,T] i32
  float* out      = (float*)d_out;           // scalar f32 loss

  const int N = in_sizes[0] / 64;            // 8192 for the reference shapes

  _Float16* z = (_Float16*)d_ws;                                  // N*64 f16
  float* acc  = (float*)((char*)d_ws + (size_t)N * 64 * sizeof(_Float16)); // 2N f32

  // Kernel 1: one wave per row.
  {
    const int threads = 256;
    const int blocks  = (N * 32 + threads - 1) / threads;
    normalize_f16_kernel<<<blocks, threads, 0, stream>>>(x, z, acc, N);
  }

  // Kernel 2: grid = (row tiles, 4); 4 waves/block -> 16 J-chunks.
  {
    const int numJT = N / 16;              // 512
    const int tilesPerChunk = numJT / 16;  // 32
    dim3 grid(N / 16, 4);
    sim_expsum_kernel<<<grid, 128, 0, stream>>>(z, tok, acc, N, tilesPerChunk);
  }

  // Kernel 3: final scalar.
  finalize_kernel<<<1, 256, 0, stream>>>(acc, out, N);
}
